// FusedAttention_49323404427917
// MI455X (gfx1250) — matrix operand order repaired and checked by reading, not compiled, not run on hardware
//
#include <hip/hip_runtime.h>
#include <cstdint>

typedef __attribute__((ext_vector_type(16))) _Float16 v16h;
typedef __attribute__((ext_vector_type(8)))  _Float16 v8h;
typedef __attribute__((ext_vector_type(8)))  float    v8f;

constexpr int Bc   = 2;
constexpr int Tc   = 2048;
constexpr int Dm   = 2048;
constexpr int Hn   = 16;
constexpr int HDim = 128;
constexpr int HALF = 64;
constexpr int Ec   = 3 * Dm; // 6144

// ---------------------------------------------------------------------------
// WMMA fragment helpers (CDNA5 16x16x32 f16, wave32)
// A-matrix 16x32: lane L -> row M=L%16; v16h elems [0..7]=K(kb..kb+7),
// [8..15]=K(kb+16..kb+23), kb=(L>>4)*8.
// B-matrix 32x16: lane L -> col N=L%16; v16h elems same K order as A.
// C/D 16x16 f32:  lane L -> col N=L%16; VGPR r -> row M=r+8*(L>>4).
// ---------------------------------------------------------------------------
__device__ __forceinline__ v16h load_a_frag(const _Float16* rowPtr, int kOff, int lane) {
    const int kb = (lane >> 4) * 8;
    v8h lo = *(const v8h*)(rowPtr + kOff + kb);
    v8h hi = *(const v8h*)(rowPtr + kOff + kb + 16);
    return __builtin_shufflevector(lo, hi, 0,1,2,3,4,5,6,7,8,9,10,11,12,13,14,15);
}

__device__ __forceinline__ v16h load_b_frag(const _Float16* colPtr, int kOff, int lane) {
    const int kb = (lane >> 4) * 8;
    v8h lo = *(const v8h*)(colPtr + kOff + kb);
    v8h hi = *(const v8h*)(colPtr + kOff + kb + 16);
    return __builtin_shufflevector(lo, hi, 0,1,2,3,4,5,6,7,8,9,10,11,12,13,14,15);
}

__device__ __forceinline__ v8f wmma16(v16h a, v16h b, v8f c) {
    return __builtin_amdgcn_wmma_f32_16x16x32_f16(false, a, false, b, (short)0, c, false, false);
}

// Async global->LDS copy of 16 bytes (per lane), tracked by ASYNCcnt.
__device__ __forceinline__ void async_copy_b128(uint32_t ldsOff, const _Float16* gptr) {
    unsigned long long ga = (unsigned long long)(uintptr_t)gptr;
    asm volatile("global_load_async_to_lds_b128 %0, %1, off"
                 :: "v"(ldsOff), "v"(ga) : "memory");
}

// ---------------------------------------------------------------------------
// Kernel 0: fp32 -> f16 conversion (x, w_qkv, w_out)
// ---------------------------------------------------------------------------
__global__ void cvt_f32_to_f16(const float* __restrict__ in, _Float16* __restrict__ out, int n) {
    int i = blockIdx.x * blockDim.x + threadIdx.x;
    const int stride = gridDim.x * blockDim.x;
    for (; i < n; i += stride) out[i] = (_Float16)in[i];
}

// ---------------------------------------------------------------------------
// Staged GEMM geometry (shared by QKV and out-proj):
//   block tile 128(M) x 64(N), K-step 64, double-buffered LDS staging via
//   GLOBAL_LOAD_ASYNC_TO_LDS_B128.  8 waves: wm=w&3 (32 rows), wn=w>>2
//   (32 cols) -> per wave 2 A-frags x 2 B-frags, 8 WMMAs per k-iteration.
// LDS rows padded to 72 f16 (144B) to spread banks.
// ---------------------------------------------------------------------------
constexpr int AP     = 72;                 // padded LDS row length in f16
constexpr int KSTEP  = 64;
constexpr int KITERS = Dm / KSTEP;         // 32
constexpr int ABYTES = 128 * AP * 2;       // 18432
constexpr int BBYTES = 64  * AP * 2;       // 9216
constexpr int STAGEB = ABYTES + BBYTES;    // 27648 per buffer

// Stage one K-step of A (128x64 f16) and B (64x64 f16) into LDS buffers.
// 1536 16B chunks / 256 threads = 6 async ops per thread (6 per wave on ASYNCcnt).
__device__ __forceinline__ void stage_tiles(const _Float16* Abase, const _Float16* Bbase,
                                            int kOff, char* bufA, char* bufB, int tid) {
#pragma unroll
    for (int i = 0; i < 4; ++i) {                 // A: 1024 chunks
        const int c   = tid + i * 256;
        const int row = c >> 3;
        const int sub = c & 7;
        const _Float16* gp = Abase + (size_t)row * Dm + kOff + sub * 8;
        const uint32_t  la = (uint32_t)(uintptr_t)(bufA + row * (AP * 2) + sub * 16);
        async_copy_b128(la, gp);
    }
#pragma unroll
    for (int i = 0; i < 2; ++i) {                 // B: 512 chunks
        const int c   = tid + i * 256;
        const int row = c >> 3;
        const int sub = c & 7;
        const _Float16* gp = Bbase + (size_t)row * Dm + kOff + sub * 8;
        const uint32_t  la = (uint32_t)(uintptr_t)(bufB + row * (AP * 2) + sub * 16);
        async_copy_b128(la, gp);
    }
}

// ---------------------------------------------------------------------------
// Kernel 1: QKV GEMM + bias + clip, scatter pre-RoPE to Q/K/V [B][H][T][HD].
// grid = (M/128=32, E/64=96)
// ---------------------------------------------------------------------------
__global__ __launch_bounds__(256) void qkv_gemm_kernel(
    const _Float16* __restrict__ xh, const _Float16* __restrict__ wh,
    const float* __restrict__ bqkv,
    _Float16* __restrict__ Qb, _Float16* __restrict__ Kb, _Float16* __restrict__ Vb)
{
    __shared__ __align__(16) char pool[2 * STAGEB];

    const int tid  = threadIdx.x;
    const int lane = tid & 31;
    const int w    = tid >> 5;
    const int n0   = lane & 15;
    const int hi   = lane >> 4;
    const int wm   = w & 3;
    const int wn   = w >> 2;

    const int mBase = blockIdx.x * 128;
    const int eBase = blockIdx.y * 64;
    const _Float16* Abase = xh + (size_t)mBase * Dm;
    const _Float16* Bbase = wh + (size_t)eBase * Dm;

    v8f acc[2][2] = {};

    stage_tiles(Abase, Bbase, 0, pool, pool + ABYTES, tid);

    for (int ks = 0; ks < KITERS; ++ks) {
        __syncthreads();   // buffer (ks+1)&1 free to overwrite
        if (ks + 1 < KITERS) {
            char* buf = pool + ((ks + 1) & 1) * STAGEB;
            stage_tiles(Abase, Bbase, (ks + 1) * KSTEP, buf, buf + ABYTES, tid);
            asm volatile("s_wait_asynccnt 0x6" ::: "memory");   // stage ks done
        } else {
            asm volatile("s_wait_asynccnt 0x0" ::: "memory");
        }
        __syncthreads();   // stage ks visible to all waves

        const _Float16* As = (const _Float16*)(pool + (ks & 1) * STAGEB);
        const _Float16* Bs = (const _Float16*)(pool + (ks & 1) * STAGEB + ABYTES);
#pragma unroll
        for (int kc = 0; kc < 2; ++kc) {
            v16h af[2];
#pragma unroll
            for (int a = 0; a < 2; ++a)
                af[a] = load_a_frag(As + (wm * 32 + a * 16 + n0) * AP, kc * 32, lane);
#pragma unroll
            for (int ns = 0; ns < 2; ++ns) {
                v16h bf = load_b_frag(Bs + (wn * 32 + ns * 16 + n0) * AP, kc * 32, lane);
#pragma unroll
                for (int a = 0; a < 2; ++a)
                    acc[a][ns] = wmma16(af[a], bf, acc[a][ns]);
            }
        }
    }

    // epilogue: bias + clip(+-8), scatter to Q/K/V (pre-RoPE)
    const int p = eBase >> 11;          // 0=q,1=k,2=v
    const int h = (eBase >> 7) & 15;
    _Float16* dstT = (p == 0) ? Qb : (p == 1) ? Kb : Vb;
#pragma unroll
    for (int ns = 0; ns < 2; ++ns) {
        const int col  = wn * 32 + ns * 16 + n0;
        const float bias = bqkv[eBase + col];
        const int d = (eBase & 127) + col;     // 0..127 within head
#pragma unroll
        for (int a = 0; a < 2; ++a) {
#pragma unroll
            for (int r = 0; r < 8; ++r) {
                const int m  = mBase + wm * 32 + a * 16 + r + 8 * hi;
                const int bI = m >> 11;
                const int t  = m & (Tc - 1);
                float v = acc[a][ns][r] + bias;
                v = fminf(8.0f, fmaxf(-8.0f, v));
                dstT[(((size_t)(bI * Hn + h)) * Tc + t) * HDim + d] = (_Float16)v;
            }
        }
    }
}

// ---------------------------------------------------------------------------
// Kernel 2: RoPE applied in place to Q and K (f16, [B][H][T][HD]).
// idx enumerates (b,h,t,d<64) pairs.
// ---------------------------------------------------------------------------
__global__ __launch_bounds__(256) void rope_kernel(
    _Float16* __restrict__ Qb, _Float16* __restrict__ Kb,
    const float* __restrict__ pos_sin, const float* __restrict__ pos_cos)
{
    const int total = Bc * Hn * Tc * HALF;   // 4,194,304
    int idx = blockIdx.x * blockDim.x + threadIdx.x;
    const int stride = gridDim.x * blockDim.x;
    for (; idx < total; idx += stride) {
        const int d  = idx & (HALF - 1);
        const int t  = (idx >> 6) & (Tc - 1);
        const int bh = idx >> 17;
        const size_t base = ((size_t)bh * Tc + t) * HDim;
        const float cv = pos_cos[t * HALF + d];
        const float sv = pos_sin[t * HALF + d];
        {
            const float t1 = (float)Qb[base + d];
            const float t2 = (float)Qb[base + d + HALF];
            Qb[base + d]        = (_Float16)(t1 * cv - t2 * sv);
            Qb[base + d + HALF] = (_Float16)(t1 * sv + t2 * cv);
        }
        {
            const float t1 = (float)Kb[base + d];
            const float t2 = (float)Kb[base + d + HALF];
            Kb[base + d]        = (_Float16)(t1 * cv - t2 * sv);
            Kb[base + d + HALF] = (_Float16)(t1 * sv + t2 * cv);
        }
    }
}

// ---------------------------------------------------------------------------
// Kernel 3: flash attention (causal), BM=64 query rows, BN=64 key rows.
// Block = 256 threads = 8 waves: wave w -> wm=w&3 (16-row slab), wn=w>>2
// (column half for S and d-half for O). Output written as f16 att[B][T][D].
// grid = (T/64=32, H=16, B=2)
// ---------------------------------------------------------------------------
__global__ __launch_bounds__(256) void flash_kernel(
    const _Float16* __restrict__ Qb, const _Float16* __restrict__ Kb,
    const _Float16* __restrict__ Vb, _Float16* __restrict__ att)
{
    constexpr float SCALE = 0.08838834764831845f; // 1/sqrt(128)

    __shared__ __align__(32) float    Ss[64 * 65];     // S scores (fp32)
    __shared__ __align__(32) _Float16 Ps[64 * 72];     // probs (f16, A-frag layout)
    __shared__ __align__(32) _Float16 Vt[128 * 80];    // V^T: [d][k]
    __shared__ float mState[64], lState[64], sState[64];

    const int tid  = threadIdx.x;
    const int lane = tid & 31;
    const int w    = tid >> 5;
    const int n0   = lane & 15;
    const int hi   = lane >> 4;
    const int wm   = w & 3;       // row slab
    const int wn   = w >> 2;      // column half

    const int bI    = blockIdx.z;
    const int h     = blockIdx.y;
    const int iBase = blockIdx.x * 64;

    const _Float16* Qh = Qb + ((size_t)(bI * Hn + h)) * Tc * HDim;
    const _Float16* Kh = Kb + ((size_t)(bI * Hn + h)) * Tc * HDim;
    const _Float16* Vh = Vb + ((size_t)(bI * Hn + h)) * Tc * HDim;

    // Preload Q A-fragments (16 rows x HD=128 -> 4 k-steps)
    const int qRow = iBase + wm * 16 + n0;
    v16h qf[4];
#pragma unroll
    for (int ks = 0; ks < 4; ++ks)
        qf[ks] = load_a_frag(Qh + (size_t)qRow * HDim, ks * 32, lane);

    v8f o[4] = {};   // 16 rows x 64 cols (this wave's d-half)

    if (tid < 64) { mState[tid] = -1e30f; lState[tid] = 0.0f; }
    __syncthreads();

    const int jEnd = blockIdx.x;            // causal: key blocks 0..iBlock
    for (int j = 0; j <= jEnd; ++j) {
        const int jBase = j * 64;

        // Stage V^T into LDS: Vt[d][k] = V[jBase+k][d]
        for (int idx = tid; idx < 64 * 128; idx += 256) {
            const int d  = idx & 127;
            const int kk = idx >> 7;
            Vt[d * 80 + kk] = Vh[(size_t)(jBase + kk) * HDim + d];
        }

        // S = Q K^T (this wave: cols [wn*32, wn*32+32))
        const bool needMask = (jBase + 63) > (iBase + wm * 16);
#pragma unroll
        for (int s2 = 0; s2 < 2; ++s2) {
            const int ns2 = wn * 2 + s2;
            v8f s = {};
            const _Float16* kPtr = Kh + (size_t)(jBase + ns2 * 16 + n0) * HDim;
#pragma unroll
            for (int ks = 0; ks < 4; ++ks) {
                v16h bf = load_b_frag(kPtr, ks * 32, lane);
                s = wmma16(qf[ks], bf, s);
            }
            const int nIdx = jBase + ns2 * 16 + n0;
#pragma unroll
            for (int r = 0; r < 8; ++r) {
                const int row  = wm * 16 + r + 8 * hi;
                const int mIdx = iBase + row;
                float v = s[r] * SCALE;
                if (needMask && nIdx > mIdx) v = -1e30f;
                Ss[row * 65 + ns2 * 16 + n0] = v;
            }
        }
        __syncthreads();

        // Online softmax: one thread per row
        if (tid < 64) {
            const int r = tid;
            const float mOld = mState[r];
            float mx = mOld;
            for (int c = 0; c < 64; ++c) mx = fmaxf(mx, Ss[r * 65 + c]);
            const float sc = __expf(mOld - mx);
            float sum = 0.0f;
            for (int c = 0; c < 64; ++c) {
                const float pv = __expf(Ss[r * 65 + c] - mx);
                sum += pv;
                Ps[r * 72 + c] = (_Float16)pv;
            }
            mState[r] = mx;
            lState[r] = lState[r] * sc + sum;
            sState[r] = sc;
        }
        __syncthreads();

        // Rescale O accumulators by per-row correction factor
        {
            const int rb = wm * 16 + 8 * hi;
            float scl[8];
#pragma unroll
            for (int r = 0; r < 8; ++r) scl[r] = sState[rb + r];
#pragma unroll
            for (int ns = 0; ns < 4; ++ns)
#pragma unroll
                for (int r = 0; r < 8; ++r) o[ns][r] *= scl[r];
        }

        // O += P * V   (A=P from LDS, B=V^T from LDS)
        v16h pf[2];
#pragma unroll
        for (int ks = 0; ks < 2; ++ks)
            pf[ks] = load_a_frag(Ps + (size_t)(wm * 16 + n0) * 72, ks * 32, lane);
#pragma unroll
        for (int ns = 0; ns < 4; ++ns) {
            const _Float16* vPtr = Vt + (size_t)(wn * 64 + ns * 16 + n0) * 80;
#pragma unroll
            for (int ks = 0; ks < 2; ++ks) {
                v16h bf = load_b_frag(vPtr, ks * 32, lane);
                o[ns] = wmma16(pf[ks], bf, o[ns]);
            }
        }
        __syncthreads();
    }

    // Normalize and store f16 att[B][T][D]
    const int rb = wm * 16 + 8 * hi;
#pragma unroll
    for (int r = 0; r < 8; ++r) {
        const float linv = 1.0f / lState[rb + r];
        const int t = iBase + rb + r;
        _Float16* dst = att + ((size_t)(bI * Tc + t)) * Dm + h * HDim + wn * 64;
#pragma unroll
        for (int ns = 0; ns < 4; ++ns)
            dst[ns * 16 + n0] = (_Float16)(o[ns][r] * linv);
    }
}

// ---------------------------------------------------------------------------
// Kernel 4: out projection (staged GEMM), fp32 output. grid = (32, 32)
// ---------------------------------------------------------------------------
__global__ __launch_bounds__(256) void outproj_kernel(
    const _Float16* __restrict__ ah, const _Float16* __restrict__ wh,
    const float* __restrict__ bout, float* __restrict__ out)
{
    __shared__ __align__(16) char pool[2 * STAGEB];

    const int tid  = threadIdx.x;
    const int lane = tid & 31;
    const int w    = tid >> 5;
    const int n0   = lane & 15;
    const int hi   = lane >> 4;
    const int wm   = w & 3;
    const int wn   = w >> 2;

    const int mBase = blockIdx.x * 128;
    const int eBase = blockIdx.y * 64;
    const _Float16* Abase = ah + (size_t)mBase * Dm;
    const _Float16* Bbase = wh + (size_t)eBase * Dm;

    v8f acc[2][2] = {};

    stage_tiles(Abase, Bbase, 0, pool, pool + ABYTES, tid);

    for (int ks = 0; ks < KITERS; ++ks) {
        __syncthreads();
        if (ks + 1 < KITERS) {
            char* buf = pool + ((ks + 1) & 1) * STAGEB;
            stage_tiles(Abase, Bbase, (ks + 1) * KSTEP, buf, buf + ABYTES, tid);
            asm volatile("s_wait_asynccnt 0x6" ::: "memory");
        } else {
            asm volatile("s_wait_asynccnt 0x0" ::: "memory");
        }
        __syncthreads();

        const _Float16* As = (const _Float16*)(pool + (ks & 1) * STAGEB);
        const _Float16* Bs = (const _Float16*)(pool + (ks & 1) * STAGEB + ABYTES);
#pragma unroll
        for (int kc = 0; kc < 2; ++kc) {
            v16h af[2];
#pragma unroll
            for (int a = 0; a < 2; ++a)
                af[a] = load_a_frag(As + (wm * 32 + a * 16 + n0) * AP, kc * 32, lane);
#pragma unroll
            for (int ns = 0; ns < 2; ++ns) {
                v16h bf = load_b_frag(Bs + (wn * 32 + ns * 16 + n0) * AP, kc * 32, lane);
#pragma unroll
                for (int a = 0; a < 2; ++a)
                    acc[a][ns] = wmma16(af[a], bf, acc[a][ns]);
            }
        }
    }

#pragma unroll
    for (int ns = 0; ns < 2; ++ns) {
        const int col  = wn * 32 + ns * 16 + n0;
        const float bias = bout[eBase + col];
#pragma unroll
        for (int a = 0; a < 2; ++a) {
#pragma unroll
            for (int r = 0; r < 8; ++r) {
                const int m = mBase + wm * 32 + a * 16 + r + 8 * hi;
                out[(size_t)m * Dm + eBase + col] = acc[a][ns][r] + bias;
            }
        }
    }
}

// ---------------------------------------------------------------------------
// Host launcher
// ---------------------------------------------------------------------------
extern "C" void kernel_launch(void* const* d_in, const int* in_sizes, int n_in,
                              void* d_out, int out_size, void* d_ws, size_t ws_size,
                              hipStream_t stream) {
    (void)in_sizes; (void)n_in; (void)out_size; (void)ws_size;

    const float* x       = (const float*)d_in[0];
    const float* w_qkv   = (const float*)d_in[1];
    const float* b_qkv   = (const float*)d_in[2];
    const float* w_out   = (const float*)d_in[3];
    const float* b_out   = (const float*)d_in[4];
    const float* pos_sin = (const float*)d_in[5];
    const float* pos_cos = (const float*)d_in[6];
    float* out = (float*)d_out;

    _Float16* ws = (_Float16*)d_ws;
    size_t off = 0;
    const size_t N_X   = (size_t)Bc * Tc * Dm;        // 8,388,608
    const size_t N_WQ  = (size_t)Ec * Dm;             // 12,582,912
    const size_t N_WO  = (size_t)Dm * Dm;             // 4,194,304
    const size_t N_QKV = (size_t)Bc * Hn * Tc * HDim; // 8,388,608

    _Float16* xh    = ws + off; off += N_X;
    _Float16* wqkvh = ws + off; off += N_WQ;
    _Float16* wouth = ws + off; off += N_WO;
    _Float16* Qb    = ws + off; off += N_QKV;
    _Float16* Kb    = ws + off; off += N_QKV;
    _Float16* Vb    = ws + off; off += N_QKV;
    _Float16* att   = xh;   // xh is dead after the QKV GEMM; reuse for att

    cvt_f32_to_f16<<<4096, 256, 0, stream>>>(x, xh, (int)N_X);
    cvt_f32_to_f16<<<4096, 256, 0, stream>>>(w_qkv, wqkvh, (int)N_WQ);
    cvt_f32_to_f16<<<2048, 256, 0, stream>>>(w_out, wouth, (int)N_WO);

    qkv_gemm_kernel<<<dim3(32, 96), 256, 0, stream>>>(xh, wqkvh, b_qkv, Qb, Kb, Vb);

    rope_kernel<<<4096, 256, 0, stream>>>(Qb, Kb, pos_sin, pos_cos);

    flash_kernel<<<dim3(Tc / 64, Hn, Bc), 256, 0, stream>>>(Qb, Kb, Vb, att);

    outproj_kernel<<<dim3(32, 32), 256, 0, stream>>>(att, wouth, b_out, out);
}
